// ExpertChoiceLayer_25228637897300
// MI455X (gfx1250) — matrix operand order repaired and checked by reading, not compiled, not run on hardware
//
#include <hip/hip_runtime.h>
#include <math.h>

// Problem constants (from reference): B=8, T=4096, D=512, E=64, H=1024, CAP=512
#define NTOK 32768
#define DIMD 512
#define NEXP 64
#define HID  1024
#define CAPC 512

typedef __attribute__((ext_vector_type(16))) __bf16 v16bf;
typedef __attribute__((ext_vector_type(8)))  float  v8f;

// ---------- CDNA5 async global->LDS copy (probe-guarded) ----------

#if defined(__has_builtin)
#if __has_builtin(__builtin_amdgcn_global_load_async_to_lds_b128)
#define HAVE_ASYNC_COPY 1
#endif
#endif

// exact parameter type from the compiler diagnostic: vector_size(16) int
typedef int v4i __attribute__((vector_size(16)));

// integer-route address-space casts (always compile; AS3 ptr = low 32 bits)
#define GPTR(p) ((__attribute__((address_space(1))) v4i*)(unsigned long long)(p))
#define LPTR(p) ((__attribute__((address_space(3))) v4i*)(unsigned)(unsigned long long)(p))

__device__ inline void async_wait0() {
#if defined(HAVE_ASYNC_COPY)
#if __has_builtin(__builtin_amdgcn_s_wait_asynccnt)
  __builtin_amdgcn_s_wait_asynccnt(0);
#else
  asm volatile("s_wait_asynccnt 0x0" ::: "memory");
#endif
#endif
}

// ---------- helpers ----------

__device__ inline unsigned short f2bf(float f) {
  unsigned u = __builtin_bit_cast(unsigned, f);
  u += 0x7FFFu + ((u >> 16) & 1u);          // round to nearest even
  return (unsigned short)(u >> 16);
}

// Build a 16-element bf16 fragment from two 8-element contiguous LDS chunks.
__device__ inline v16bf make_frag16(const unsigned short* p0, const unsigned short* p1) {
  v16bf r;
#pragma unroll
  for (int i = 0; i < 8; ++i) {
    r[i]     = __builtin_bit_cast(__bf16, p0[i]);
    r[i + 8] = __builtin_bit_cast(__bf16, p1[i]);
  }
  return r;
}

// monotonic key for float ordering (ascending)
__device__ inline unsigned fkey(float f) {
  unsigned u = __builtin_bit_cast(unsigned, f);
  return u ^ ((u & 0x80000000u) ? 0xFFFFFFFFu : 0x80000000u);
}

// ---------- kernel 0: init output + constant aux stats ----------

__global__ __launch_bounds__(256) void init_out_kernel(float* __restrict__ out) {
  size_t i = (size_t)blockIdx.x * 256 + threadIdx.x;
  const size_t total = (size_t)NTOK * DIMD;
  if (i < total) out[i] = 0.0f;
  if (i == 0) {
    const float frac = (float)CAPC / (float)NTOK;  // each expert picks exactly CAP tokens
    out[total]     = frac * frac;                  // aux_loss = mean(frac^2), all equal
    out[total + 1] = frac;                         // mean_cap_util
  }
}

// ---------- kernel 1: router logits (writes transposed (E, N)) ----------

__global__ __launch_bounds__(256) void router_kernel(const float* __restrict__ x,
                                                     const float* __restrict__ rw,
                                                     float* __restrict__ logitsT) {
  int idx = blockIdx.x * 256 + threadIdx.x;   // N*16 threads, 4 experts each
  int n  = idx >> 4;
  int eg = idx & 15;
  const float* xr = x + (size_t)n * DIMD;
  const float* wp = rw + eg * 4;
  float a0 = 0.f, a1 = 0.f, a2 = 0.f, a3 = 0.f;
  for (int k = 0; k < DIMD; ++k) {
    float xv = xr[k];
    const float* w4 = wp + (size_t)k * NEXP;
    a0 += xv * w4[0]; a1 += xv * w4[1]; a2 += xv * w4[2]; a3 += xv * w4[3];
  }
  int e0 = eg * 4;
  logitsT[(size_t)(e0 + 0) * NTOK + n] = a0;
  logitsT[(size_t)(e0 + 1) * NTOK + n] = a1;
  logitsT[(size_t)(e0 + 2) * NTOK + n] = a2;
  logitsT[(size_t)(e0 + 3) * NTOK + n] = a3;
}

// ---------- kernel 2: per-expert softmax stats over tokens ----------

__global__ __launch_bounds__(256) void stats_kernel(const float* __restrict__ logitsT,
                                                    float* __restrict__ smax,
                                                    float* __restrict__ ssum) {
  const int e = blockIdx.x, tid = threadIdx.x;
  const float* row = logitsT + (size_t)e * NTOK;
  __shared__ float red[256];
  float m = -3.4e38f;
  for (int n = tid; n < NTOK; n += 256) m = fmaxf(m, row[n]);
  red[tid] = m; __syncthreads();
  for (int s = 128; s; s >>= 1) { if (tid < s) red[tid] = fmaxf(red[tid], red[tid + s]); __syncthreads(); }
  m = red[0]; __syncthreads();
  float acc = 0.f;
  for (int n = tid; n < NTOK; n += 256) acc += __expf(row[n] - m);
  red[tid] = acc; __syncthreads();
  for (int s = 128; s; s >>= 1) { if (tid < s) red[tid] += red[tid + s]; __syncthreads(); }
  if (tid == 0) { smax[e] = m; ssum[e] = red[0]; }
}

// ---------- kernel 3: per-expert top-CAP radix select ----------

__global__ __launch_bounds__(256) void select_kernel(const float* __restrict__ logitsT,
                                                     const float* __restrict__ smax,
                                                     const float* __restrict__ ssum,
                                                     int* __restrict__ tidxo,
                                                     float* __restrict__ ewo) {
  const int e = blockIdx.x;
  const float* row = logitsT + (size_t)e * NTOK;
  __shared__ unsigned hist[256];
  __shared__ unsigned s_prefix, s_K, s_gt, s_eq;
  const int tid = threadIdx.x;
  if (tid == 0) { s_prefix = 0u; s_K = CAPC; }
  unsigned high_mask = 0u;
  for (int pass = 0; pass < 4; ++pass) {
    const int shift = 24 - 8 * pass;
    hist[tid] = 0u;
    __syncthreads();
    const unsigned pref = s_prefix;
    for (int n = tid; n < NTOK; n += 256) {
      unsigned u = fkey(row[n]);
      if ((u & high_mask) == pref)
        atomicAdd(&hist[(u >> shift) & 255u], 1u);
    }
    __syncthreads();
    if (tid == 0) {
      unsigned need = s_K, accum = 0u;
      int b = 255;
      for (; b >= 0; --b) {
        unsigned c = hist[b];
        if (accum + c >= need) break;
        accum += c;
      }
      if (b < 0) b = 0;
      s_prefix |= ((unsigned)b) << shift;
      s_K = need - accum;
    }
    high_mask |= (0xFFu << shift);
    __syncthreads();
  }
  const unsigned T = s_prefix;        // exact threshold key
  const unsigned need_eq = s_K;       // # of ties to take
  const unsigned base_eq = CAPC - need_eq;
  if (tid == 0) { s_gt = 0u; s_eq = 0u; }
  __syncthreads();
  const float m = smax[e], s = ssum[e];
  for (int n = tid; n < NTOK; n += 256) {
    float v = row[n];
    unsigned u = fkey(v);
    int slot = -1;
    if (u > T) {
      slot = (int)atomicAdd(&s_gt, 1u);
    } else if (u == T) {
      unsigned t = atomicAdd(&s_eq, 1u);
      if (t < need_eq) slot = (int)(base_eq + t);
    }
    if (slot >= 0) {
      tidxo[(size_t)e * CAPC + slot] = n;
      ewo  [(size_t)e * CAPC + slot] = __expf(v - m) / s;   // softmax prob (expert weight)
    }
  }
}

// ---------- kernel 4: gather selected tokens -> bf16 A (E, CAP, D) ----------

__global__ __launch_bounds__(128) void gather_kernel(const float* __restrict__ x,
                                                     const int* __restrict__ tidx,
                                                     unsigned short* __restrict__ Abf) {
  const int rowi = blockIdx.x;                  // 0 .. E*CAP-1
  const int tok = tidx[rowi];
  const float* src = x + (size_t)tok * DIMD;
  unsigned short* dst = Abf + (size_t)rowi * DIMD;
  for (int i = threadIdx.x; i < DIMD; i += 128) dst[i] = f2bf(src[i]);
}

// ---------- shared GEMM mainloop (double-buffered, async A staging) ----------
// Block tile 128(M) x 128(N), 8 waves arranged 2(M) x 4(N): wave tile 64x32.
// A: bf16 global (rows x K) -> LDS via GLOBAL_LOAD_ASYNC_TO_LDS_B128 (ASYNCcnt).
// B: fp32 global (K x ldB), converted to bf16 and stored transposed in LDS so
// fragment loads are contiguous per lane. LDS rows padded to 40 ushorts.
// Two LDS buffers: stage tile t+1 while computing tile t.

#define TILE_USHORTS (128 * 40)

__device__ inline void stage_tiles(const unsigned short* __restrict__ Ag, int ldA,
                                   const float* __restrict__ Bg, int ldB,
                                   int k0,
                                   unsigned short* ldsA, unsigned short* ldsB) {
  const int tid = threadIdx.x;
  // A tile 128x32 bf16 (straight copy, 2 x 16B per thread)
  {
    const int row = tid >> 1, col = (tid & 1) * 16;
    const unsigned short* sp = Ag + (size_t)row * ldA + k0 + col;
    unsigned short* dp = &ldsA[row * 40 + col];
#if defined(HAVE_ASYNC_COPY)
    __builtin_amdgcn_global_load_async_to_lds_b128(GPTR(sp),     LPTR(dp),     0, 0);
    __builtin_amdgcn_global_load_async_to_lds_b128(GPTR(sp + 8), LPTR(dp + 8), 0, 0);
#else
    uint4 d0 = *(const uint4*)sp;
    uint4 d1 = *(const uint4*)(sp + 8);
    *(uint4*)dp       = d0;
    *(uint4*)(dp + 8) = d1;
#endif
  }
  // B tile 32x128: fp32 -> bf16, transposed into ldsB[n][k]
  {
    const int n = tid & 127, kb = (tid >> 7) * 16;
    alignas(16) unsigned short tmp[16];
#pragma unroll
    for (int i = 0; i < 16; ++i)
      tmp[i] = f2bf(Bg[(size_t)(k0 + kb + i) * ldB + n]);
    *(uint4*)&ldsB[n * 40 + kb]     = ((const uint4*)tmp)[0];
    *(uint4*)&ldsB[n * 40 + kb + 8] = ((const uint4*)tmp)[1];
  }
}

__device__ inline void gemm_mainloop(const unsigned short* __restrict__ Ag, int ldA,
                                     const float* __restrict__ Bg, int ldB,
                                     int K, v8f acc[8],
                                     unsigned short* ldsA, unsigned short* ldsB) {
  const int tid   = threadIdx.x;
  const int lane  = tid & 31, w = tid >> 5;
  const int waveM = w >> 2, waveN = w & 3;
  const int r = lane & 15, half = lane >> 4;
  const int T = K >> 5;

  // prologue: stage tile 0 into buffer 0
  stage_tiles(Ag, ldA, Bg, ldB, 0, ldsA, ldsB);
  async_wait0();
  __syncthreads();

  for (int t = 0; t < T; ++t) {
    unsigned short* cA = ldsA + (t & 1) * TILE_USHORTS;
    unsigned short* cB = ldsB + (t & 1) * TILE_USHORTS;

    if (t + 1 < T) {
      stage_tiles(Ag, ldA, Bg, ldB, (t + 1) << 5,
                  ldsA + ((t + 1) & 1) * TILE_USHORTS,
                  ldsB + ((t + 1) & 1) * TILE_USHORTS);
      if (t + 2 < T)  // hint tile t+2's B into cache (global_prefetch_b8)
        __builtin_prefetch(&Bg[(size_t)((t + 2) << 5) * ldB + (tid & 127)], 0, 1);
    }

    // fragments. A 16x32: lanes 0-15 rows M, elems 0..7 = K0..7, 8..15 = K16..23;
    // lanes 16-31: K8..15 / K24..31.  B 32x16 (stored [n][k]): lane r = col,
    // elems in the same K order as A.
    v16bf afr[4], bfr[2];
#pragma unroll
    for (int f = 0; f < 4; ++f) {
      const unsigned short* p = &cA[(waveM * 64 + f * 16 + r) * 40];
      afr[f] = make_frag16(p + half * 8, p + 16 + half * 8);
    }
#pragma unroll
    for (int g = 0; g < 2; ++g) {
      const unsigned short* p = &cB[(waveN * 32 + g * 16 + r) * 40];
      bfr[g] = make_frag16(p + half * 8, p + 16 + half * 8);
    }
#pragma unroll
    for (int f = 0; f < 4; ++f)
#pragma unroll
      for (int g = 0; g < 2; ++g)
        acc[f * 2 + g] = __builtin_amdgcn_wmma_f32_16x16x32_bf16(
            false, afr[f], false, bfr[g], (short)0, acc[f * 2 + g], false, false);

    async_wait0();     // next tile's async A copies complete
    __syncthreads();   // + everyone done with this buffer
  }
}

// ---------- kernel 5: fc1 GEMM + bias + GELU -> bf16 h (E, CAP, H) ----------

__global__ __launch_bounds__(256) void fc1_kernel(const unsigned short* __restrict__ Abf,
                                                  const float* __restrict__ w1,
                                                  const float* __restrict__ b1,
                                                  unsigned short* __restrict__ hbuf) {
  alignas(16) __shared__ unsigned short ldsA[2 * TILE_USHORTS];
  alignas(16) __shared__ unsigned short ldsB[2 * TILE_USHORTS];
  const int b  = blockIdx.x;
  const int e  = b >> 5;                 // 32 tiles per expert (4 M x 8 N)
  const int rt = b & 31;
  const int mt = rt >> 3, nt = rt & 7;

  const unsigned short* Ag = Abf + ((size_t)e * CAPC + mt * 128) * DIMD;
  const float*          Bg = w1  + (size_t)e * DIMD * HID + nt * 128;

  v8f acc[8];
#pragma unroll
  for (int i = 0; i < 8; ++i) acc[i] = (v8f){0.f, 0.f, 0.f, 0.f, 0.f, 0.f, 0.f, 0.f};

  gemm_mainloop(Ag, DIMD, Bg, HID, DIMD, acc, ldsA, ldsB);

  const int tid = threadIdx.x, lane = tid & 31, w = tid >> 5;
  const int waveM = w >> 2, waveN = w & 3, r = lane & 15, half = lane >> 4;
#pragma unroll
  for (int f = 0; f < 4; ++f) {
#pragma unroll
    for (int g = 0; g < 2; ++g) {
      const int n = nt * 128 + waveN * 32 + g * 16 + r;
      const float bias = b1[(size_t)e * HID + n];
      v8f a = acc[f * 2 + g];
#pragma unroll
      for (int j = 0; j < 8; ++j) {
        const int m = mt * 128 + waveM * 64 + f * 16 + half * 8 + j;
        float v = a[j] + bias;
        float ge = 0.5f * v * (1.0f + erff(v * 0.70710678118654752f));  // exact GELU
        hbuf[((size_t)e * CAPC + m) * HID + n] = f2bf(ge);
      }
    }
  }
}

// ---------- kernel 6: fc2 GEMM + bias, weighted atomic scatter-add ----------

__global__ __launch_bounds__(256) void fc2_kernel(const unsigned short* __restrict__ hbuf,
                                                  const float* __restrict__ w2,
                                                  const float* __restrict__ b2,
                                                  const int* __restrict__ tidx,
                                                  const float* __restrict__ ew,
                                                  float* __restrict__ out) {
  alignas(16) __shared__ unsigned short ldsA[2 * TILE_USHORTS];
  alignas(16) __shared__ unsigned short ldsB[2 * TILE_USHORTS];
  const int b  = blockIdx.x;
  const int e  = b >> 4;                 // 16 tiles per expert (4 M x 4 N)
  const int rt = b & 15;
  const int mt = rt >> 2, nt = rt & 3;

  const unsigned short* Ag = hbuf + ((size_t)e * CAPC + mt * 128) * HID;
  const float*          Bg = w2   + (size_t)e * HID * DIMD + nt * 128;

  v8f acc[8];
#pragma unroll
  for (int i = 0; i < 8; ++i) acc[i] = (v8f){0.f, 0.f, 0.f, 0.f, 0.f, 0.f, 0.f, 0.f};

  gemm_mainloop(Ag, HID, Bg, DIMD, HID, acc, ldsA, ldsB);

  const int tid = threadIdx.x, lane = tid & 31, w = tid >> 5;
  const int waveM = w >> 2, waveN = w & 3, r = lane & 15, half = lane >> 4;
#pragma unroll
  for (int f = 0; f < 4; ++f) {
#pragma unroll
    for (int g = 0; g < 2; ++g) {
      const int n = nt * 128 + waveN * 32 + g * 16 + r;
      const float bias = b2[(size_t)e * DIMD + n];
      v8f a = acc[f * 2 + g];
#pragma unroll
      for (int j = 0; j < 8; ++j) {
        const int m = mt * 128 + waveM * 64 + f * 16 + half * 8 + j;
        const int   tok = tidx[(size_t)e * CAPC + m];
        const float wgt = ew  [(size_t)e * CAPC + m];
        atomicAdd(&out[(size_t)tok * DIMD + n], (a[j] + bias) * wgt);
      }
    }
  }
}

// ---------- launch ----------

extern "C" void kernel_launch(void* const* d_in, const int* in_sizes, int n_in,
                              void* d_out, int out_size, void* d_ws, size_t ws_size,
                              hipStream_t stream) {
  (void)in_sizes; (void)n_in; (void)out_size; (void)ws_size;
  const float* x  = (const float*)d_in[0];
  const float* rw = (const float*)d_in[1];
  const float* w1 = (const float*)d_in[2];
  const float* b1 = (const float*)d_in[3];
  const float* w2 = (const float*)d_in[4];
  const float* b2 = (const float*)d_in[5];
  float* out = (float*)d_out;

  char* ws = (char*)d_ws;
  float*          logitsT = (float*)         (ws + 0);         // E*N*4      = 8388608
  float*          smax    = (float*)         (ws + 8388608);   // E*4 (pad)
  float*          ssum    = (float*)         (ws + 8388864);
  int*            tidx    = (int*)           (ws + 8389120);   // E*CAP*4    = 131072
  float*          ew      = (float*)         (ws + 8520192);   // E*CAP*4    = 131072
  unsigned short* Abf     = (unsigned short*)(ws + 8651264);   // E*CAP*D*2  = 33554432
  unsigned short* hbuf    = (unsigned short*)(ws + 42205696);  // E*CAP*H*2  = 67108864
                                                               // total ~104 MB

  init_out_kernel<<<(NTOK * DIMD) / 256, 256, 0, stream>>>(out);
  router_kernel  <<<(NTOK * 16) / 256,   256, 0, stream>>>(x, rw, logitsT);
  stats_kernel   <<<NEXP,                256, 0, stream>>>(logitsT, smax, ssum);
  select_kernel  <<<NEXP,                256, 0, stream>>>(logitsT, smax, ssum, tidx, ew);
  gather_kernel  <<<NEXP * CAPC,         128, 0, stream>>>(x, tidx, Abf);
  fc1_kernel     <<<NEXP * 32,           256, 0, stream>>>(Abf, w1, b1, hbuf);
  fc2_kernel     <<<NEXP * 16,           256, 0, stream>>>(hbuf, w2, b2, tidx, ew, out);
}
